// EncoderNew_31834297598147
// MI455X (gfx1250) — compile-verified, matrix operand order checked by reading, not run on hardware
//
#include <hip/hip_runtime.h>

typedef _Float16 h16;
typedef _Float16 v16h __attribute__((ext_vector_type(16)));
typedef float    v8f  __attribute__((ext_vector_type(8)));

#define BDIM 16
#define SDIM 512
#define CDIM 16
#define DDIM 512
#define FDIM 2048
#define SELD 64
#define MROWS (BDIM*SDIM)   /* 8192 */
#define LDA 40              /* padded LDS row stride (halves); 40*2=80 bytes, 16B aligned */

// ---------------------------------------------------------------------------
// CDNA5 async direct-to-LDS copy (CDNA5 ISA): per-lane 16B
// transfer global->LDS, tracked by ASYNCcnt. Inline asm (portable across
// ROCm7.2 / amdgpu-toolchain builtin arity differences).
// ---------------------------------------------------------------------------
static __device__ __forceinline__ void async_b128(unsigned lds_byte, const void* gaddr) {
  asm volatile("global_load_async_to_lds_b128 %0, %1, off"
               :: "v"(lds_byte), "v"(gaddr) : "memory");
}
static __device__ __forceinline__ void wait_async0() {
  asm volatile("s_wait_asynccnt 0" ::: "memory");
}

// ---------------------------------------------------------------------------
// Fragment load following the CDNA5 16-bit A/B layout (CDNA5 ISA §7.12.2):
//   lanes 0-15 : rows 0-15, halves 0..7 -> K=0..7,  halves 8..15 -> K=16..23
//   lanes 16-31: rows 0-15, halves 0..7 -> K=8..15, halves 8..15 -> K=24..31
// r = lane&15, koff = (lane>>4)*8. Both 8-half chunks contiguous -> b128.
// ---------------------------------------------------------------------------
static __device__ __forceinline__ v16h load_frag(const h16* base, int ld, int r, int koff) {
  v16h v;
  const h16* p0 = base + r * ld + koff;
#pragma unroll
  for (int i = 0; i < 8; ++i) { v[i] = p0[i]; v[i + 8] = p0[16 + i]; }
  return v;
}

static __device__ __forceinline__ v8f wmma_f16(v16h a, v16h b, v8f c) {
  return __builtin_amdgcn_wmma_f32_16x16x32_f16(false, a, false, b, (short)0, c, false, false);
}

static __device__ __forceinline__ v8f zero8() {
  v8f z;
#pragma unroll
  for (int i = 0; i < 8; ++i) z[i] = 0.0f;
  return z;
}

// ---------------------------------------------------------------------------
// 1) Fused embedding: sum_c emb[code]*mask_code + bias + time-MLP + pos-enc
// ---------------------------------------------------------------------------
__global__ __launch_bounds__(256) void embed_kernel(
    const int* __restrict__ event_code, const float* __restrict__ mask_code,
    const float* __restrict__ seq_time, const int* __restrict__ input_len,
    const float* __restrict__ emb, const float* __restrict__ bias_emb,
    const float* __restrict__ selW, const float* __restrict__ selB,
    const float* __restrict__ timeW, const float* __restrict__ timeB,
    float* __restrict__ out32, h16* __restrict__ out16) {
  int row = blockIdx.x;                  // b*S + s
  int b = row >> 9, s = row & 511;
  __shared__ float hid[SELD];
  __shared__ int   codes[CDIM];
  __shared__ float mc[CDIM];
  int tid = threadIdx.x;
  if (tid < SELD) {
    float t = seq_time[row] * (1.0f / 7200.0f);
    float x = t * selW[tid] + selB[tid];
    hid[tid] = 1.0f - tanhf(x * x);
  }
  if (tid < CDIM) {
    codes[tid] = event_code[row * CDIM + tid];
    mc[tid]    = mask_code[row * CDIM + tid];
  }
  __syncthreads();
  bool valid = (s + 1) <= input_len[b];
  for (int d = tid; d < DDIM; d += 256) {
    float acc = bias_emb[d];
#pragma unroll 4
    for (int c = 0; c < CDIM; ++c) acc += emb[(size_t)codes[c] * DDIM + d] * mc[c];
    float tf = timeB[d];
#pragma unroll 8
    for (int j = 0; j < SELD; ++j) tf += hid[j] * timeW[j * DDIM + d];
    float pe = 0.0f;
    if (valid) {
      float e   = (float)(2 * (d >> 1)) * (1.0f / (float)DDIM);
      float ang = (float)s * __expf(-e * 9.210340372f);   // 10000^-e
      pe = (d & 1) ? __cosf(ang) : __sinf(ang);
    }
    float v = acc + tf + pe;
    out32[(size_t)row * DDIM + d] = v;
    out16[(size_t)row * DDIM + d] = (h16)v;
  }
}

// ---------------------------------------------------------------------------
// 2) f32 -> f16 weight conversion WITH transpose: src[K,N] -> dst[N,K].
//    One-time cost; makes every GEMM tile-stage a contiguous async b128.
// ---------------------------------------------------------------------------
__global__ void convert_transpose(const float* __restrict__ src, h16* __restrict__ dst,
                                  int K, int N) {
  int i = blockIdx.x * blockDim.x + threadIdx.x;
  if (i >= K * N) return;
  int k = i / N, n = i - k * N;
  dst[(size_t)n * K + k] = (h16)src[i];
}

// ---------------------------------------------------------------------------
// 3) WMMA GEMM: out = act(A[M,K](f16) @ Wt[N,K](f16,row=out-col) + bias [+res])
//    Block tile 128x64, 8 waves, double-buffered LDS fed by async b128 copies;
//    next K-step's tiles are issued before computing the current step.
//    Compile-time fusion flags -> branch-free epilogue.
//    Assumes M%128==0, N%64==0, K%32==0 (true at all call sites).
// ---------------------------------------------------------------------------
template <bool RELU, bool RESID, bool OUT32, bool OUT16>
__global__ __launch_bounds__(256) void gemm_wmma(
    const h16* __restrict__ A, const h16* __restrict__ Wt,
    const float* __restrict__ bias, const float* __restrict__ resid,
    float* __restrict__ out32, h16* __restrict__ out16,
    int M, int N, int K) {
  __shared__ __align__(16) h16 ldsA[2][128 * LDA];
  __shared__ __align__(16) h16 ldsW[2][64 * LDA];
  int tid = threadIdx.x;
  int wave = tid >> 5, lane = tid & 31;
  int m = lane & 15, koff = (lane >> 4) << 3;
  int rowBase = blockIdx.y * 128;
  int colBase = blockIdx.x * 64;

  // per-thread staging assignment (all 16B-aligned, contiguous chunks)
  int ar = tid >> 1, ac = (tid & 1) << 4;         // A: 128 rows x 32 halves
  int wn = tid >> 2, wk = (tid & 3) << 3;         // Wt: 64 rows x 32 halves
  const h16* gA = A + (size_t)(rowBase + ar) * K + ac;
  const h16* gW = Wt + (size_t)(colBase + wn) * K + wk;
  unsigned lA = (unsigned)(size_t)&ldsA[0][ar * LDA + ac];
  unsigned lW = (unsigned)(size_t)&ldsW[0][wn * LDA + wk];
  unsigned bufStrideA = (unsigned)(sizeof(h16) * 128 * LDA);
  unsigned bufStrideW = (unsigned)(sizeof(h16) * 64 * LDA);

  auto issue = [&](int k0, int p) {
    async_b128(lA + p * bufStrideA,      gA + k0);
    async_b128(lA + p * bufStrideA + 16, gA + k0 + 8);
    async_b128(lW + p * bufStrideW,      gW + k0);
  };

  v8f acc[4];
#pragma unroll
  for (int t = 0; t < 4; ++t) acc[t] = zero8();

  issue(0, 0);
  int p = 0;
  for (int k0 = 0; k0 < K; k0 += 32) {
    wait_async0();          // this wave's async tiles landed in LDS
    __syncthreads();        // all waves' tiles landed; prev compute finished
    if (k0 + 32 < K) issue(k0 + 32, p ^ 1);   // prefetch next step
    v16h af = load_frag(&ldsA[p][(wave << 4) * LDA], LDA, m, koff);
#pragma unroll
    for (int t = 0; t < 4; ++t) {
      v16h bf = load_frag(&ldsW[p][(t << 4) * LDA], LDA, m, koff);
      acc[t] = wmma_f16(af, bf, acc[t]);
    }
    p ^= 1;
  }

  int hi8 = (lane >> 4) << 3, col0 = lane & 15;
#pragma unroll
  for (int t = 0; t < 4; ++t) {
    int col = colBase + (t << 4) + col0;
    float bv = bias[col];
#pragma unroll
    for (int r = 0; r < 8; ++r) {
      int row = rowBase + (wave << 4) + r + hi8;   // C layout: vgpr r -> M=r / r+8
      float v = acc[t][r] + bv;
      if (RELU) v = fmaxf(v, 0.0f);
      size_t idx = (size_t)row * N + col;
      if (RESID) v += resid[idx];
      if (OUT32) out32[idx] = v;
      if (OUT16) out16[idx] = (h16)v;
    }
  }
}

// ---------------------------------------------------------------------------
// 4) V transpose: contiguous [128][512][64] -> Vt [128][64][512]
// ---------------------------------------------------------------------------
__global__ void transpose_v(const h16* __restrict__ v, h16* __restrict__ vt, int total) {
  int i = blockIdx.x * blockDim.x + threadIdx.x;
  if (i >= total) return;
  int bb = i >> 15, rem = i & 32767;
  int ss = rem >> 6, d = rem & 63;
  vt[((size_t)bb << 15) + d * 512 + ss] = v[i];
}

// ---------------------------------------------------------------------------
// 5) Fused WMMA attention. Q/K/V as contiguous [128][512][64] (the raw
//    reshape in the reference is exactly this reinterpretation).
//    One block = (bb, 16 query rows): WMMA scores -> LDS softmax -> WMMA P@V.
// ---------------------------------------------------------------------------
__global__ __launch_bounds__(128) void attn_wmma(
    const h16* __restrict__ Q, const h16* __restrict__ Km,
    const h16* __restrict__ Vt, h16* __restrict__ Ctx) {
  __shared__ float sc[16 * 512];   // 32 KB score tile
  __shared__ h16   pb[16 * 512];   // 16 KB f16 probabilities
  int qt = blockIdx.x;             // 0..31
  int bb = blockIdx.y;             // 0..127
  int tid = threadIdx.x;
  int wave = tid >> 5, lane = tid & 31;
  int m = lane & 15, koff = (lane >> 4) << 3, hi8 = (lane >> 4) << 3;
  const h16* qbase = Q + ((size_t)bb << 15) + qt * 16 * 64;
  const h16* kbase = Km + ((size_t)bb << 15);

  // scores: Q[16x64] @ K^T ; B-fragment = K rows directly (contiguous)
  for (int kt = wave * 8; kt < wave * 8 + 8; ++kt) {
    v8f acc = zero8();
    const h16* kb = kbase + kt * 16 * 64;
#pragma unroll
    for (int kk = 0; kk < 64; kk += 32) {
      v16h af = load_frag(qbase + kk, 64, m, koff);
      v16h bf = load_frag(kb + kk, 64, m, koff);
      acc = wmma_f16(af, bf, acc);
    }
#pragma unroll
    for (int r = 0; r < 8; ++r)
      sc[(r + hi8) * 512 + kt * 16 + (lane & 15)] = acc[r] * 0.125f;  // dh^-0.5
  }
  __syncthreads();

  // softmax per row: 8 lanes/row, shuffle reductions inside wave32
  {
    int row = tid >> 3, sub = tid & 7;
    float mx = -3.0e38f;
    for (int j = sub; j < 512; j += 8) mx = fmaxf(mx, sc[row * 512 + j]);
#pragma unroll
    for (int o = 4; o >= 1; o >>= 1) mx = fmaxf(mx, __shfl_xor(mx, o, 32));
    float sum = 0.0f;
    for (int j = sub; j < 512; j += 8) {
      float e = __expf(sc[row * 512 + j] - mx);
      sc[row * 512 + j] = e;
      sum += e;
    }
#pragma unroll
    for (int o = 4; o >= 1; o >>= 1) sum += __shfl_xor(sum, o, 32);
    float inv = 1.0f / sum;
    for (int j = sub; j < 512; j += 8) pb[row * 512 + j] = (h16)(sc[row * 512 + j] * inv);
  }
  __syncthreads();

  // ctx = P[16x512] @ V[512x64]; B-fragment from Vt rows (contiguous)
  {
    int cb = wave << 4;   // each wave owns 16 of the 64 dh columns
    v8f acc = zero8();
    const h16* vb = Vt + ((size_t)bb << 15) + (size_t)cb * 512;
    for (int ks = 0; ks < 512; ks += 32) {
      v16h af = load_frag(pb + ks, 512, m, koff);
      v16h bf = load_frag(vb + ks, 512, m, koff);
      acc = wmma_f16(af, bf, acc);
    }
#pragma unroll
    for (int r = 0; r < 8; ++r) {
      int row = qt * 16 + r + hi8;
      Ctx[((size_t)bb << 15) + (size_t)row * 64 + cb + (lane & 15)] = (h16)acc[r];
    }
  }
}

// ---------------------------------------------------------------------------
// 6) LayerNorm over D=512, one block per row, dual f32/f16 output
// ---------------------------------------------------------------------------
__global__ __launch_bounds__(256) void layernorm_kernel(
    const float* __restrict__ x, const float* __restrict__ g, const float* __restrict__ bta,
    float eps, float* __restrict__ out32, h16* __restrict__ out16) {
  __shared__ float red[256];
  int row = blockIdx.x, tid = threadIdx.x;
  const float* xr = x + (size_t)row * DDIM;
  float a = xr[tid], b = xr[tid + 256];
  red[tid] = a + b;
  __syncthreads();
  for (int o = 128; o > 0; o >>= 1) { if (tid < o) red[tid] += red[tid + o]; __syncthreads(); }
  float mu = red[0] * (1.0f / DDIM);
  __syncthreads();
  float da = a - mu, db = b - mu;
  red[tid] = da * da + db * db;
  __syncthreads();
  for (int o = 128; o > 0; o >>= 1) { if (tid < o) red[tid] += red[tid + o]; __syncthreads(); }
  float rstd = rsqrtf(red[0] * (1.0f / DDIM) + eps);
  float y0 = da * rstd * g[tid] + bta[tid];
  float y1 = db * rstd * g[tid + 256] + bta[tid + 256];
  out32[(size_t)row * DDIM + tid]       = y0;
  out32[(size_t)row * DDIM + tid + 256] = y1;
  out16[(size_t)row * DDIM + tid]       = (h16)y0;
  out16[(size_t)row * DDIM + tid + 256] = (h16)y1;
}

// ---------------------------------------------------------------------------
extern "C" void kernel_launch(void* const* d_in, const int* in_sizes, int n_in,
                              void* d_out, int out_size, void* d_ws, size_t ws_size,
                              hipStream_t stream) {
  (void)in_sizes; (void)n_in; (void)out_size; (void)ws_size;
  const int*   event_code = (const int*)d_in[0];
  const float* mask_code  = (const float*)d_in[2];
  const float* seq_time   = (const float*)d_in[3];
  const int*   input_len  = (const int*)d_in[6];
  const float* emb        = (const float*)d_in[7];
  const float* bias_emb   = (const float*)d_in[8];
  const float* selW = (const float*)d_in[9];  const float* selB = (const float*)d_in[10];
  const float* timeW = (const float*)d_in[11]; const float* timeB = (const float*)d_in[12];
  const float* Wq = (const float*)d_in[13]; const float* bq = (const float*)d_in[14];
  const float* Wk = (const float*)d_in[15]; const float* bk = (const float*)d_in[16];
  const float* Wv = (const float*)d_in[17]; const float* bv = (const float*)d_in[18];
  const float* Wo = (const float*)d_in[19]; const float* bo = (const float*)d_in[20];
  const float* ln1g = (const float*)d_in[21]; const float* ln1b = (const float*)d_in[22];
  const float* W1 = (const float*)d_in[23]; const float* b1 = (const float*)d_in[24];
  const float* W2 = (const float*)d_in[25]; const float* b2 = (const float*)d_in[26];
  const float* ln2g = (const float*)d_in[27]; const float* ln2b = (const float*)d_in[28];
  float* outp = (float*)d_out;

  // workspace layout (bytes); h1 aliases the q/k/v/vt region (dead by FFN time)
  char* ws = (char*)d_ws;
  h16*   q16   = (h16*)(ws + 0);
  h16*   k16   = (h16*)(ws + 8388608);
  h16*   v16   = (h16*)(ws + 16777216);
  h16*   vt16  = (h16*)(ws + 25165824);
  h16*   ctx16 = (h16*)(ws + 33554432);
  h16*   h116  = (h16*)(ws + 0);
  h16*   act16 = (h16*)(ws + 41943040);
  float* act32 = (float*)(ws + 50331648);
  float* tmp32 = (float*)(ws + 67108864);
  h16*   wq16 = (h16*)(ws + 83886080);   // all weight copies are TRANSPOSED [N,K]
  h16*   wk16 = wq16 + DDIM * DDIM;
  h16*   wv16 = wk16 + DDIM * DDIM;
  h16*   wo16 = wv16 + DDIM * DDIM;
  h16*   w116 = wo16 + DDIM * DDIM;
  h16*   w216 = w116 + DDIM * FDIM;

  embed_kernel<<<MROWS, 256, 0, stream>>>(event_code, mask_code, seq_time, input_len,
                                          emb, bias_emb, selW, selB, timeW, timeB,
                                          act32, act16);

  for (int l = 0; l < 2; ++l) {
    int oDD = l * DDIM * DDIM, oDF = l * DDIM * FDIM, oD = l * DDIM, oF = l * FDIM;
    convert_transpose<<<(DDIM * DDIM) / 256, 256, 0, stream>>>(Wq + oDD, wq16, DDIM, DDIM);
    convert_transpose<<<(DDIM * DDIM) / 256, 256, 0, stream>>>(Wk + oDD, wk16, DDIM, DDIM);
    convert_transpose<<<(DDIM * DDIM) / 256, 256, 0, stream>>>(Wv + oDD, wv16, DDIM, DDIM);
    convert_transpose<<<(DDIM * DDIM) / 256, 256, 0, stream>>>(Wo + oDD, wo16, DDIM, DDIM);
    convert_transpose<<<(DDIM * FDIM) / 256, 256, 0, stream>>>(W1 + oDF, w116, DDIM, FDIM);
    convert_transpose<<<(DDIM * FDIM) / 256, 256, 0, stream>>>(W2 + oDF, w216, FDIM, DDIM);

    // QKV projections (f16 out for attention)
    gemm_wmma<false, false, false, true><<<dim3(DDIM / 64, MROWS / 128), 256, 0, stream>>>(
        act16, wq16, bq + oD, nullptr, nullptr, q16, MROWS, DDIM, DDIM);
    gemm_wmma<false, false, false, true><<<dim3(DDIM / 64, MROWS / 128), 256, 0, stream>>>(
        act16, wk16, bk + oD, nullptr, nullptr, k16, MROWS, DDIM, DDIM);
    gemm_wmma<false, false, false, true><<<dim3(DDIM / 64, MROWS / 128), 256, 0, stream>>>(
        act16, wv16, bv + oD, nullptr, nullptr, v16, MROWS, DDIM, DDIM);

    transpose_v<<<(MROWS * DDIM) / 256, 256, 0, stream>>>(v16, vt16, MROWS * DDIM);
    attn_wmma<<<dim3(32, 128), 128, 0, stream>>>(q16, k16, vt16, ctx16);

    // Wo projection + residual, then LN1
    gemm_wmma<false, true, true, false><<<dim3(DDIM / 64, MROWS / 128), 256, 0, stream>>>(
        ctx16, wo16, bo + oD, act32, tmp32, nullptr, MROWS, DDIM, DDIM);
    layernorm_kernel<<<MROWS, 256, 0, stream>>>(tmp32, ln1g + oD, ln1b + oD, 1e-3f,
                                                act32, act16);

    // FFN: relu(X @ W1 + b1) @ W2 + b2 + residual, then LN2
    gemm_wmma<true, false, false, true><<<dim3(FDIM / 64, MROWS / 128), 256, 0, stream>>>(
        act16, w116, b1 + oF, nullptr, nullptr, h116, MROWS, FDIM, DDIM);
    gemm_wmma<false, true, true, false><<<dim3(DDIM / 64, MROWS / 128), 256, 0, stream>>>(
        h116, w216, b2 + oD, act32, tmp32, nullptr, MROWS, DDIM, FDIM);
    float* oo = (l == 1) ? outp : act32;
    layernorm_kernel<<<MROWS, 256, 0, stream>>>(tmp32, ln2g + oD, ln2b + oD, 1e-6f,
                                                oo, act16);
  }
}
